// GRNNEncoder_69200513073744
// MI455X (gfx1250) — matrix operand order repaired and checked by reading, not compiled, not run on hardware
//
#include <hip/hip_runtime.h>
#include <hip/hip_bf16.h>
#include <cstdint>

// ---------------------------------------------------------------------------
// NRI GRNN encoder on MI455X (gfx1250).
// All dense layers run through a bf16 WMMA GEMM (v_wmma_f32_16x16x32_bf16,
// f32 accumulate). Weights converted to bf16 once per launch; activations
// converted to bf16 while staging into LDS with full 128-bit vector paths.
// Edge gathers (node2edge concat) are fused into the GEMM A-tile load.
// Block tile 128x64, 8 waves, each wave a 32x32 micro-tile (4 WMMA / K-step,
// 2 A-frags + 2 B-frags -> 2x matrix-op per LDS byte vs 16x32 slabs).
// GEMM invariants: M % 128 == 0, Nc % 64 == 0, K % 32 == 0 (MLP1's K=4 is
// zero-padded to 32 up front), all row strides 16B-aligned.
// ---------------------------------------------------------------------------

typedef __attribute__((ext_vector_type(16))) __bf16 v16bf;
typedef __attribute__((ext_vector_type(8)))  float  v8f;

constexpr int HD = 256;      // hidden
constexpr int NN = 16;       // nodes
constexpr int NE = 240;      // edges
constexpr int TS = 64;       // timesteps
constexpr int NBATCH = 8;    // batch
constexpr int BT = NBATCH * TS;          // 512
constexpr int ROWS_NODE = BT * NN;       // 8192
constexpr int ROWS_EDGE = BT * NE;       // 122880
constexpr int LSTM_B = NBATCH * NE;      // 1920
constexpr float BN_EPS = 1e-5f;

__device__ __forceinline__ unsigned int f2bf1(float f) {
    unsigned int u = __float_as_uint(f);
    unsigned int r = u + 0x7FFFu + ((u >> 16) & 1u);   // round-to-nearest-even
    return r >> 16;
}
__device__ __forceinline__ unsigned int packbf(float lo, float hi) {
    return f2bf1(lo) | (f2bf1(hi) << 16);
}
__device__ __forceinline__ float elu1(float x) {
    return x > 0.f ? x : (__expf(x) - 1.f);
}
__device__ __forceinline__ float sigm(float x) {
    return 1.f / (1.f + __expf(-x));
}

__device__ __forceinline__ uint4 pack8(const float* __restrict__ src) {
    float4 f0 = *(const float4*)src;
    float4 f1 = *(const float4*)(src + 4);
    uint4 pk;
    pk.x = packbf(f0.x, f0.y);
    pk.y = packbf(f0.z, f0.w);
    pk.z = packbf(f1.x, f1.y);
    pk.w = packbf(f1.z, f1.w);
    return pk;
}

// ---------------------------------------------------------------------------
// WMMA GEMM: C[M,Nc] = act( A[M,K] * W[Nc,K]^T + bias (+C if accumulate) )
// MODE 0: A0[aBase + row*lda + k]
// MODE 2: edge concat:   k<256 -> node[send], else node[recv]    (K=512)
// MODE 3: edge concat3:  k<256 send, k<512 recv, else skip[row]  (K=768)
// ---------------------------------------------------------------------------
template <int MODE>
__global__ __launch_bounds__(256) void gemm_wmma_bf16(
    const float* __restrict__ A0, const float* __restrict__ A1,
    const unsigned short* __restrict__ W, const float* __restrict__ bias,
    float* __restrict__ C,
    const int* __restrict__ sendI, const int* __restrict__ recvI,
    int M, int Nc, int K, long aBase, int lda, int act, int accumulate)
{
    constexpr int LDT = 40;  // ushort stride, 80B rows (16B aligned, depadded)
    __shared__ __align__(16) unsigned short As[128 * LDT];  // 10 KB
    __shared__ __align__(16) unsigned short Ws[64 * LDT];   //  5 KB

    const int tid  = threadIdx.x;
    const int m0   = blockIdx.x * 128;
    const int n0   = blockIdx.y * 64;
    const int lane = tid & 31;
    const int wave = tid >> 5;
    const int wr   = wave >> 1;   // 0..3  (32-row slab)
    const int wc   = wave & 1;    // 0..1  (32-col slab)
    const int hh   = lane >> 4;   // lane half
    const int ln   = lane & 15;

    // A staging: one row, 16 consecutive k per thread (2x uint4)
    const int sra  = tid >> 1;          // 0..127
    const int ska  = (tid & 1) * 16;    // 0 or 16
    const int grow = m0 + sra;
    // W staging: one row, 8 consecutive k per thread (1x uint4)
    const int srw  = tid >> 2;          // 0..63
    const int skw  = (tid & 3) * 8;     // 0,8,16,24

    long rowbase0 = 0, rowbase1 = 0, rowbase2 = 0;
    if (MODE == 0) {
        rowbase0 = aBase + (long)grow * lda;
    } else {
        int bt = grow / NE, e = grow - bt * NE;
        int s = sendI[e], r = recvI[e];
        rowbase0 = ((long)(bt * NN + s)) * HD;
        rowbase1 = ((long)(bt * NN + r)) * HD;
        if (MODE == 3) rowbase2 = (long)grow * HD;
    }
    const unsigned short* wrow = W + (long)(n0 + srw) * K;

    v8f acc00 = {}, acc01 = {}, acc10 = {}, acc11 = {};

    for (int k0 = 0; k0 < K; k0 += 32) {
        // ---- stage A (128x32): chunk of 16 k never crosses a 256-aligned
        // gather segment (ska/k0 are 16/32 aligned, segments 256-aligned)
        const int gka = k0 + ska;
        const float* src;
        if (MODE == 0) {
            src = A0 + rowbase0 + gka;
        } else if (MODE == 2) {
            src = (gka < HD) ? (A0 + rowbase0 + gka)
                             : (A0 + rowbase1 + (gka - HD));
        } else {
            src = (gka < HD)      ? (A0 + rowbase0 + gka)
                : (gka < 2 * HD)  ? (A0 + rowbase1 + (gka - HD))
                                  : (A1 + rowbase2 + (gka - 2 * HD));
        }
        *(uint4*)&As[sra * LDT + ska]     = pack8(src);
        *(uint4*)&As[sra * LDT + ska + 8] = pack8(src + 8);
        // ---- stage W (64x32), already bf16 in memory
        *(uint4*)&Ws[srw * LDT + skw] = *(const uint4*)&wrow[k0 + skw];

        if (k0 + 32 < K)  // hint next W tile into cache (global_prefetch_b8)
            __builtin_prefetch(&wrow[k0 + 32 + skw], 0, 1);
        __syncthreads();

        // A fragments (16x32): lane half h, row m: k = [8h..8h+7], [16+8h..]
        union { v16bf v; uint4 q[2]; } af0, af1, bf0, bf1;
        const uint4* ap0 = (const uint4*)&As[(wr * 32 + ln) * LDT];
        af0.q[0] = ap0[hh];
        af0.q[1] = ap0[2 + hh];
        const uint4* ap1 = (const uint4*)&As[(wr * 32 + 16 + ln) * LDT];
        af1.q[0] = ap1[hh];
        af1.q[1] = ap1[2 + hh];
        // B fragments (32x16): lane col n: k = [8h..8h+7], [16+8h..]
        const uint4* bp0 = (const uint4*)&Ws[(wc * 32 + ln) * LDT];
        bf0.q[0] = bp0[hh];
        bf0.q[1] = bp0[2 + hh];
        const uint4* bp1 = (const uint4*)&Ws[(wc * 32 + 16 + ln) * LDT];
        bf1.q[0] = bp1[hh];
        bf1.q[1] = bp1[2 + hh];

        acc00 = __builtin_amdgcn_wmma_f32_16x16x32_bf16(
            false, af0.v, false, bf0.v, (short)0, acc00, false, false);
        acc01 = __builtin_amdgcn_wmma_f32_16x16x32_bf16(
            false, af0.v, false, bf1.v, (short)0, acc01, false, false);
        acc10 = __builtin_amdgcn_wmma_f32_16x16x32_bf16(
            false, af1.v, false, bf0.v, (short)0, acc10, false, false);
        acc11 = __builtin_amdgcn_wmma_f32_16x16x32_bf16(
            false, af1.v, false, bf1.v, (short)0, acc11, false, false);
        __syncthreads();
    }

    // epilogue: C layout — lane n = ln, VGPR r -> row r + 8*hh
    const v8f* accs[4] = {&acc00, &acc01, &acc10, &acc11};
#pragma unroll
    for (int mt = 0; mt < 2; ++mt) {
#pragma unroll
        for (int nt = 0; nt < 2; ++nt) {
            const v8f& a = *accs[mt * 2 + nt];
            int col = n0 + wc * 32 + nt * 16 + ln;
#pragma unroll
            for (int r = 0; r < 8; ++r) {
                int rowg = m0 + wr * 32 + mt * 16 + r + 8 * hh;
                float v = a[r];
                if (bias) v += bias[col];
                long o = (long)rowg * Nc + col;
                if (accumulate) v += C[o];
                if (act) v = elu1(v);
                C[o] = v;
            }
        }
    }
}

// ---------------------------------------------------------------------------
// helpers
// ---------------------------------------------------------------------------
__global__ void cvt_bf16_k(const float* __restrict__ src,
                           unsigned short* __restrict__ dst, long n) {
    long i = (long)blockIdx.x * blockDim.x + threadIdx.x;
    if (i < n) dst[i] = (unsigned short)f2bf1(src[i]);
}

// convert [rows,Ksrc] f32 row-major -> [rows,Kdst] bf16, zero-padding K
__global__ void cvt_pad_bf16_k(const float* __restrict__ src,
                               unsigned short* __restrict__ dst,
                               int rows, int Ksrc, int Kdst) {
    long i = (long)blockIdx.x * blockDim.x + threadIdx.x;
    long total = (long)rows * Kdst;
    if (i >= total) return;
    int k = (int)(i % Kdst);
    int r = (int)(i / Kdst);
    dst[i] = (k < Ksrc) ? (unsigned short)f2bf1(src[(long)r * Ksrc + k]) : 0;
}

// permute inputs [B,N,T,D=4] -> Apad [bt*16+n, 32] f32 (zero-padded K)
__global__ void pad_input_k(const float* __restrict__ inputs,
                            float* __restrict__ Apad) {
    long i = (long)blockIdx.x * blockDim.x + threadIdx.x;  // rows*32
    if (i >= (long)ROWS_NODE * 32) return;
    int k = (int)(i & 31);
    long row = i >> 5;
    int n = (int)(row & 15);
    int bt = (int)(row >> 4);
    int t = bt & 63, b = bt >> 6;
    Apad[i] = (k < 4) ? inputs[(((long)(b * NN + n)) * TS + t) * 4 + k] : 0.f;
}

__global__ void addv_k(const float* __restrict__ a, const float* __restrict__ b,
                       float* __restrict__ c, int n) {
    int i = blockIdx.x * blockDim.x + threadIdx.x;
    if (i < n) c[i] = a[i] + b[i];
}

__global__ void zero_k(float* __restrict__ p, long n) {
    long i = (long)blockIdx.x * blockDim.x + threadIdx.x;
    if (i < n) p[i] = 0.f;
}

__global__ void build_idx_k(const float* __restrict__ rr,
                            const float* __restrict__ rs,
                            int* __restrict__ sendI, int* __restrict__ recvI,
                            int* __restrict__ incI) {
    int tid = threadIdx.x;
    if (tid < NE) {
        int s = 0, r = 0;
        for (int n = 0; n < NN; ++n) {
            if (rs[tid * NN + n] > 0.5f) s = n;
            if (rr[tid * NN + n] > 0.5f) r = n;
        }
        sendI[tid] = s;
        recvI[tid] = r;
    }
    __threadfence_block();
    __syncthreads();
    if (tid < NN) {
        int cnt = 0;
        for (int e = 0; e < NE; ++e)
            if (recvI[e] == tid) incI[tid * 15 + cnt++] = e;
    }
}

__global__ __launch_bounds__(256) void bn_stats_k(const float* __restrict__ X,
                                                  int rows,
                                                  float* __restrict__ meanv,
                                                  float* __restrict__ varv) {
    __shared__ float ssum[256], ssq[256];
    int c = blockIdx.x, tid = threadIdx.x;
    float s = 0.f, q = 0.f;
    for (int r = tid; r < rows; r += 256) {
        float v = X[(long)r * HD + c];
        s += v;
        q += v * v;
    }
    ssum[tid] = s;
    ssq[tid] = q;
    __syncthreads();
    for (int w = 128; w > 0; w >>= 1) {
        if (tid < w) { ssum[tid] += ssum[tid + w]; ssq[tid] += ssq[tid + w]; }
        __syncthreads();
    }
    if (tid == 0) {
        float m = ssum[0] / rows;
        meanv[c] = m;
        varv[c] = ssq[0] / rows - m * m;
    }
}

// perm==1: src row = (b*64+t)*240+e  ->  dst row = (b*240+e)*64+t
__global__ void bn_apply_k(const float* __restrict__ X, float* __restrict__ Y,
                           const float* __restrict__ meanv,
                           const float* __restrict__ varv,
                           const float* __restrict__ g,
                           const float* __restrict__ be, int rows, int perm) {
    long idx = (long)blockIdx.x * blockDim.x + threadIdx.x;
    long total = (long)rows * HD;
    if (idx >= total) return;
    int c = (int)(idx & (HD - 1));
    long r = idx >> 8;
    float v = (X[idx] - meanv[c]) * rsqrtf(varv[c] + BN_EPS) * g[c] + be[c];
    if (!perm) {
        Y[idx] = v;
    } else {
        int e = (int)(r % NE);
        int bt = (int)(r / NE);
        int t = bt & 63, b = bt >> 6;
        long dr = ((long)(b * NE + e)) * TS + t;
        Y[dr * HD + c] = v;
    }
}

__global__ __launch_bounds__(256) void edge2node_k(
    const float* __restrict__ Xe, float* __restrict__ Xn,
    const int* __restrict__ incI) {
    int blk = blockIdx.x;  // bt*16 + n
    int n = blk & 15, bt = blk >> 4;
    int c = threadIdx.x;
    float a = 0.f;
#pragma unroll
    for (int j = 0; j < 15; ++j) {
        int e = incI[n * 15 + j];
        a += Xe[((long)(bt * NE + e)) * HD + c];
    }
    Xn[(long)blk * HD + c] = a * (1.0f / 16.0f);
}

__global__ __launch_bounds__(256) void lstm_cell_k(
    const float* __restrict__ gates, float* __restrict__ hbuf,
    float* __restrict__ cbuf, float* __restrict__ seq_out, int t) {
    int r = blockIdx.x, c = threadIdx.x;
    long gb = (long)r * 1024;
    float gi = gates[gb + c];
    float gf = gates[gb + 256 + c];
    float gg = gates[gb + 512 + c];
    float go = gates[gb + 768 + c];
    long hb = (long)r * HD + c;
    float cc = sigm(gf) * cbuf[hb] + sigm(gi) * tanhf(gg);
    float hv = sigm(go) * tanhf(cc);
    cbuf[hb] = cc;
    hbuf[hb] = hv;
    seq_out[(long)r * (TS * HD) + (long)t * HD + c] = hv;
}

__global__ __launch_bounds__(256) void final_fc_k(
    const float* __restrict__ hbuf, const float* __restrict__ w,
    const float* __restrict__ b, float* __restrict__ out) {
    __shared__ float s0[256], s1[256];
    int r = blockIdx.x, c = threadIdx.x;
    float hv = hbuf[(long)r * HD + c];
    s0[c] = hv * w[c];
    s1[c] = hv * w[HD + c];
    __syncthreads();
    for (int w2 = 128; w2 > 0; w2 >>= 1) {
        if (c < w2) { s0[c] += s0[c + w2]; s1[c] += s1[c + w2]; }
        __syncthreads();
    }
    if (c == 0) {
        out[r * 2 + 0] = s0[0] + b[0];
        out[r * 2 + 1] = s1[0] + b[1];
    }
}

// ---------------------------------------------------------------------------
extern "C" void kernel_launch(void* const* d_in, const int* in_sizes, int n_in,
                              void* d_out, int out_size, void* d_ws,
                              size_t ws_size, hipStream_t stream) {
    (void)in_sizes; (void)n_in; (void)out_size; (void)ws_size;

    const float* inputs   = (const float*)d_in[0];
    const float* rel_rec  = (const float*)d_in[1];
    const float* rel_send = (const float*)d_in[2];
    const float *mw1[4], *mb1[4], *mw2[4], *mb2[4], *mg[4], *mbe[4];
    for (int i = 0; i < 4; ++i) {
        mw1[i] = (const float*)d_in[3 + i * 6 + 0];
        mb1[i] = (const float*)d_in[3 + i * 6 + 1];
        mw2[i] = (const float*)d_in[3 + i * 6 + 2];
        mb2[i] = (const float*)d_in[3 + i * 6 + 3];
        mg[i]  = (const float*)d_in[3 + i * 6 + 4];
        mbe[i] = (const float*)d_in[3 + i * 6 + 5];
    }
    const float* lstm_wih = (const float*)d_in[27];
    const float* lstm_whh = (const float*)d_in[28];
    const float* lstm_bih = (const float*)d_in[29];
    const float* lstm_bhh = (const float*)d_in[30];
    const float* fco_w    = (const float*)d_in[31];
    const float* fco_b    = (const float*)d_in[32];
    float* out = (float*)d_out;

    // workspace carve
    char* p = (char*)d_ws;
    auto alloc = [&](size_t bytes) -> void* {
        void* r = (void*)p;
        p += (bytes + 255) & ~(size_t)255;
        return r;
    };
    // bf16 weights; index 0 is m1_w1 padded K 4 -> 32
    const int wK[8] = {32, 256, 512, 256, 256, 256, 768, 256};
    unsigned short* w_bf[8];
    for (int i = 0; i < 8; ++i)
        w_bf[i] = (unsigned short*)alloc((size_t)256 * wK[i] * 2);
    unsigned short* wih_bf = (unsigned short*)alloc((size_t)3 * 1024 * 256 * 2);
    unsigned short* whh_bf = (unsigned short*)alloc((size_t)3 * 1024 * 256 * 2);
    float* bsum  = (float*)alloc(3 * 1024 * 4);
    float* meanv = (float*)alloc(256 * 4);
    float* varv  = (float*)alloc(256 * 4);
    int* sendI = (int*)alloc(NE * 4);
    int* recvI = (int*)alloc(NE * 4);
    int* incI  = (int*)alloc(NE * 4);
    float* Apad  = (float*)alloc((size_t)ROWS_NODE * 32 * 4);
    float* nodeA = (float*)alloc((size_t)ROWS_NODE * HD * 4);
    float* nodeB = (float*)alloc((size_t)ROWS_NODE * HD * 4);
    float* edgeA = (float*)alloc((size_t)ROWS_EDGE * HD * 4);  // skip / reuse
    float* edgeB = (float*)alloc((size_t)ROWS_EDGE * HD * 4);  // hidden / reuse
    float* seqA  = (float*)alloc((size_t)ROWS_EDGE * HD * 4);  // lstm sequence
    float* gates = (float*)alloc((size_t)LSTM_B * 1024 * 4);
    float* hbuf  = (float*)alloc((size_t)LSTM_B * HD * 4);
    float* cbuf  = (float*)alloc((size_t)LSTM_B * HD * 4);

    // weight conversion f32 -> bf16 (m1_w1 padded to K=32)
    cvt_pad_bf16_k<<<(256 * 32 + 255) / 256, 256, 0, stream>>>(
        mw1[0], w_bf[0], 256, 4, 32);
    const float* wsrc[8] = {nullptr, mw2[0], mw1[1], mw2[1],
                            mw1[2], mw2[2], mw1[3], mw2[3]};
    for (int i = 1; i < 8; ++i) {
        long n = 256L * wK[i];
        cvt_bf16_k<<<(int)((n + 255) / 256), 256, 0, stream>>>(wsrc[i], w_bf[i], n);
    }
    {
        long n = 3L * 1024 * 256;
        cvt_bf16_k<<<(int)((n + 255) / 256), 256, 0, stream>>>(lstm_wih, wih_bf, n);
        cvt_bf16_k<<<(int)((n + 255) / 256), 256, 0, stream>>>(lstm_whh, whh_bf, n);
    }
    addv_k<<<(3 * 1024 + 255) / 256, 256, 0, stream>>>(lstm_bih, lstm_bhh, bsum, 3 * 1024);
    build_idx_k<<<1, 256, 0, stream>>>(rel_rec, rel_send, sendI, recvI, incI);
    pad_input_k<<<(ROWS_NODE * 32 + 255) / 256, 256, 0, stream>>>(inputs, Apad);

    dim3 blk(256);
    // ---- MLP1 (nodes): [8192,32(pad)] -> [8192,256] -> [8192,256], BN
    gemm_wmma_bf16<0><<<dim3(ROWS_NODE / 128, 4), blk, 0, stream>>>(
        Apad, nullptr, w_bf[0], mb1[0], nodeB, nullptr, nullptr,
        ROWS_NODE, 256, 32, 0, 32, 1, 0);
    gemm_wmma_bf16<0><<<dim3(ROWS_NODE / 128, 4), blk, 0, stream>>>(
        nodeB, nullptr, w_bf[1], mb2[0], nodeA, nullptr, nullptr,
        ROWS_NODE, 256, 256, 0, 256, 1, 0);
    bn_stats_k<<<256, 256, 0, stream>>>(nodeA, ROWS_NODE, meanv, varv);
    bn_apply_k<<<(ROWS_NODE * HD + 255) / 256, 256, 0, stream>>>(
        nodeA, nodeA, meanv, varv, mg[0], mbe[0], ROWS_NODE, 0);

    // ---- MLP2 (edges): concat-gather [122880,512] -> 256 -> 256, BN -> skip
    gemm_wmma_bf16<2><<<dim3(ROWS_EDGE / 128, 4), blk, 0, stream>>>(
        nodeA, nullptr, w_bf[2], mb1[1], edgeB, sendI, recvI,
        ROWS_EDGE, 256, 512, 0, 0, 1, 0);
    gemm_wmma_bf16<0><<<dim3(ROWS_EDGE / 128, 4), blk, 0, stream>>>(
        edgeB, nullptr, w_bf[3], mb2[1], edgeA, nullptr, nullptr,
        ROWS_EDGE, 256, 256, 0, 256, 1, 0);
    bn_stats_k<<<256, 256, 0, stream>>>(edgeA, ROWS_EDGE, meanv, varv);
    bn_apply_k<<<(ROWS_EDGE * HD + 255) / 256, 256, 0, stream>>>(
        edgeA, edgeA, meanv, varv, mg[1], mbe[1], ROWS_EDGE, 0);

    // ---- edge2node (sum of incoming edges / 16)
    edge2node_k<<<ROWS_NODE, 256, 0, stream>>>(edgeA, nodeB, incI);

    // ---- MLP3 (nodes)
    gemm_wmma_bf16<0><<<dim3(ROWS_NODE / 128, 4), blk, 0, stream>>>(
        nodeB, nullptr, w_bf[4], mb1[2], nodeA, nullptr, nullptr,
        ROWS_NODE, 256, 256, 0, 256, 1, 0);
    gemm_wmma_bf16<0><<<dim3(ROWS_NODE / 128, 4), blk, 0, stream>>>(
        nodeA, nullptr, w_bf[5], mb2[2], nodeB, nullptr, nullptr,
        ROWS_NODE, 256, 256, 0, 256, 1, 0);
    bn_stats_k<<<256, 256, 0, stream>>>(nodeB, ROWS_NODE, meanv, varv);
    bn_apply_k<<<(ROWS_NODE * HD + 255) / 256, 256, 0, stream>>>(
        nodeB, nodeB, meanv, varv, mg[2], mbe[2], ROWS_NODE, 0);

    // ---- MLP4 (edges, concat3: send|recv|skip, K=768), BN + permute -> seqA
    gemm_wmma_bf16<3><<<dim3(ROWS_EDGE / 128, 4), blk, 0, stream>>>(
        nodeB, edgeA, w_bf[6], mb1[3], edgeB, sendI, recvI,
        ROWS_EDGE, 256, 768, 0, 0, 1, 0);
    gemm_wmma_bf16<0><<<dim3(ROWS_EDGE / 128, 4), blk, 0, stream>>>(
        edgeB, nullptr, w_bf[7], mb2[3], edgeA, nullptr, nullptr,
        ROWS_EDGE, 256, 256, 0, 256, 1, 0);
    bn_stats_k<<<256, 256, 0, stream>>>(edgeA, ROWS_EDGE, meanv, varv);
    bn_apply_k<<<(ROWS_EDGE * HD + 255) / 256, 256, 0, stream>>>(
        edgeA, seqA, meanv, varv, mg[3], mbe[3], ROWS_EDGE, 1);

    // ---- 3-layer LSTM over T=64, batch 1920
    float* seq_in = seqA;
    float* seq_out = edgeB;
    for (int l = 0; l < 3; ++l) {
        zero_k<<<(LSTM_B * HD + 255) / 256, 256, 0, stream>>>(hbuf, (long)LSTM_B * HD);
        zero_k<<<(LSTM_B * HD + 255) / 256, 256, 0, stream>>>(cbuf, (long)LSTM_B * HD);
        const unsigned short* wih = wih_bf + (size_t)l * 1024 * 256;
        const unsigned short* whh = whh_bf + (size_t)l * 1024 * 256;
        const float* bs = bsum + l * 1024;
        for (int t = 0; t < TS; ++t) {
            // gates = x_t @ Wih^T + (bih + bhh)
            gemm_wmma_bf16<0><<<dim3(LSTM_B / 128, 16), blk, 0, stream>>>(
                seq_in, nullptr, wih, bs, gates, nullptr, nullptr,
                LSTM_B, 1024, 256, (long)t * HD, TS * HD, 0, 0);
            // gates += h @ Whh^T
            gemm_wmma_bf16<0><<<dim3(LSTM_B / 128, 16), blk, 0, stream>>>(
                hbuf, nullptr, whh, nullptr, gates, nullptr, nullptr,
                LSTM_B, 1024, 256, 0, HD, 0, 1);
            lstm_cell_k<<<LSTM_B, 256, 0, stream>>>(gates, hbuf, cbuf, seq_out, t);
        }
        float* tmp = seq_in;
        seq_in = seq_out;
        seq_out = tmp;
    }

    // ---- final FC head: out[b*240+e, 0..1]
    final_fc_k<<<LSTM_B, 256, 0, stream>>>(hbuf, fco_w, fco_b, out);
}
